// TrajGRU_7052336300626
// MI455X (gfx1250) — matrix operand order repaired and checked by reading, not compiled, not run on hardware
//
#include <hip/hip_runtime.h>
#include <math.h>

// ---------------- problem constants ----------------
#define SS 10
#define BB 8
#define HH 96
#define WW 96
#define CC 8
#define FF 64
#define LL 5
#define TF 192                 // 3*F
#define KH 320                 // L*F
#define NPIX (BB*HH*WW)        // 73728

typedef __attribute__((ext_vector_type(16))) __bf16 v16bf;
typedef __attribute__((ext_vector_type(8)))  float  v8f;

__device__ __forceinline__ int clampi(int v, int lo, int hi) {
    return v < lo ? lo : (v > hi ? hi : v);
}
__device__ __forceinline__ float leaky(float x) { return x > 0.f ? x : 0.2f * x; }
__device__ __forceinline__ float sigmoidf(float x) { return 1.f / (1.f + __expf(-x)); }

// WMMA bf16 fragment helpers -------------------------------------------------
// A frag (16x32, row r = lane&15): lane holds K{kb..kb+15} with
// kb = (lane>>4)*16 -- one contiguous 32B run -> 2x ds_load_b128 from LDS.
__device__ __forceinline__ v16bf load_a_frag(const __bf16* ap) {
    v16bf a;
#pragma unroll
    for (int j = 0; j < 8; ++j) a[j] = ap[j];
#pragma unroll
    for (int j = 0; j < 8; ++j) a[8 + j] = ap[8 + j];
    return a;
}

// ---------------- utility: zero fill ----------------
__global__ void fill_zero(float* __restrict__ p, size_t n) {
    size_t i = (size_t)blockIdx.x * blockDim.x + threadIdx.x;
    if (i < n) p[i] = 0.f;
}

// ---------------- weight fragment packing ----------------
// B-fragment layout (bf16 16x16x32 WMMA): lane n (0-15) holds column n, K=0..15
// as 16 halves; lane n+16 holds column n, K=16..31.  Each fragment is stored as
// 32 lanes x 16 halves contiguous so GEMM kernels do one 32B load per lane.

// w_ret: (320,192) -> frags[kc(10)][nt(12)]
__global__ void pack_wret(const float* __restrict__ wret, __bf16* __restrict__ dst) {
    int t = blockIdx.x * blockDim.x + threadIdx.x;          // 120*32
    if (t >= 120 * 32) return;
    int f = t >> 5, lane = t & 31;
    int kc = f / 12, nt = f % 12;
    int col = lane & 15, kb = (lane >> 4) * 16;
    for (int j = 0; j < 16; ++j) {
        int k = kc * 32 + kb + j;
        dst[(size_t)f * 512 + lane * 16 + j] = (__bf16)wret[k * TF + nt * 16 + col];
    }
}

// w_h2f: (5,5,64,32) -> frags[tap(25)][kc(2)][nt(2)]
__global__ void pack_wh2f(const float* __restrict__ wh2f, __bf16* __restrict__ dst) {
    int t = blockIdx.x * blockDim.x + threadIdx.x;          // 100*32
    if (t >= 100 * 32) return;
    int f = t >> 5, lane = t & 31;
    int nt = f & 1, kc = (f >> 1) & 1, tap = f >> 2;
    int col = lane & 15, kb = (lane >> 4) * 16;
    for (int j = 0; j < 16; ++j) {
        int k = kc * 32 + kb + j;
        dst[(size_t)f * 512 + lane * 16 + j] =
            (__bf16)wh2f[(size_t)(tap * FF + k) * 32 + nt * 16 + col];
    }
}

// w_i2f: (5,5,8,32), K = tap*8+ci (200, zero-pad to 224) -> frags[c(7)][nt(2)]
__global__ void pack_wi2f(const float* __restrict__ wi2f, __bf16* __restrict__ dst) {
    int t = blockIdx.x * blockDim.x + threadIdx.x;          // 14*32
    if (t >= 14 * 32) return;
    int f = t >> 5, lane = t & 31;
    int c = f >> 1, nt = f & 1;
    int col = lane & 15, kb = (lane >> 4) * 16;
    for (int j = 0; j < 16; ++j) {
        int kx = c * 32 + kb + j;
        float v = (kx < 200) ? wi2f[(size_t)kx * 32 + nt * 16 + col] : 0.f;
        dst[(size_t)f * 512 + lane * 16 + j] = (__bf16)v;
    }
}

// w_flows: (5,5,32,10), K = tap*32+ci, N padded 10->16 -> frags[tap(25)]
__global__ void pack_wfl(const float* __restrict__ wfl, __bf16* __restrict__ dst) {
    int t = blockIdx.x * blockDim.x + threadIdx.x;          // 25*32
    if (t >= 25 * 32) return;
    int tap = t >> 5, lane = t & 31;
    int col = lane & 15, kb = (lane >> 4) * 16;
    for (int j = 0; j < 16; ++j) {
        int kk = kb + j;
        float v = (col < 10) ? wfl[(size_t)(tap * 32 + kk) * 10 + col] : 0.f;
        dst[(size_t)tap * 512 + lane * 16 + j] = (__bf16)v;
    }
}

// w_i2h: (3,3,8,192), K = tap*8+ci (72, zero-pad to 96) -> frags[kc(3)][nt(12)]
__global__ void pack_wi2h(const float* __restrict__ wi2h, __bf16* __restrict__ dst) {
    int t = blockIdx.x * blockDim.x + threadIdx.x;          // 36*32
    if (t >= 36 * 32) return;
    int f = t >> 5, lane = t & 31;
    int kc = f / 12, nt = f % 12;
    int col = lane & 15, kb = (lane >> 4) * 16;
    for (int j = 0; j < 16; ++j) {
        int kx = kc * 32 + kb + j;
        float v = (kx < 72) ? wi2h[(size_t)kx * TF + nt * 16 + col] : 0.f;
        dst[(size_t)f * 512 + lane * 16 + j] = (__bf16)v;
    }
}

// tiling shared by the conv kernels: 32 pixels (2 waves x 16) along W, grid B*H*3
#define TPX 32
#define WCOLS 36                // 32 + 4 halo (5x5)

// ---------------- i2h: 3x3 conv C=8 -> 192 via WMMA ----------------
__global__ void __launch_bounds__(64)
i2h_kernel(const float* __restrict__ x, const __bf16* __restrict__ wi2h_pk,
           const float* __restrict__ bias, float* __restrict__ out) {
    __shared__ __align__(16) __bf16 xim[TPX][96];           // 6 KB im2col rows
    int wg = blockIdx.x;
    int xseg = wg % 3;
    int rem = wg / 3;
    int y = rem % HH, b = rem / HH;
    int xbase = xseg * TPX;
    int tid = threadIdx.x;

    for (int i = tid; i < TPX * 96; i += 64) {
        int px = i / 96, kx = i % 96;
        float v = 0.f;
        if (kx < 72) {
            int tap = kx >> 3, ci = kx & 7;
            int gy = y + tap / 3 - 1, gx = xbase + px + tap % 3 - 1;
            if (gy >= 0 && gy < HH && gx >= 0 && gx < WW)
                v = x[((size_t)(b * HH + gy) * WW + gx) * CC + ci];
        }
        xim[px][kx] = (__bf16)v;
    }
    __syncthreads();

    int wv = tid >> 5, lane = tid & 31;
    int r = lane & 15, kb = (lane >> 4) * 16;
    v8f acc[12];
#pragma unroll
    for (int nt = 0; nt < 12; ++nt) acc[nt] = (v8f){};
    for (int kc = 0; kc < 3; ++kc) {
        v16bf a = load_a_frag(&xim[wv * 16 + r][kc * 32 + kb]);
#pragma unroll
        for (int nt = 0; nt < 12; ++nt) {
            v16bf bfr = *(const v16bf*)(wi2h_pk + (size_t)(kc * 12 + nt) * 512 + lane * 16);
            acc[nt] = __builtin_amdgcn_wmma_f32_16x16x32_bf16(false, a, false, bfr,
                                                              (short)0, acc[nt], false, false);
        }
    }
    int col = lane & 15, moff = (lane >> 4) * 8;
#pragma unroll
    for (int v = 0; v < 8; ++v) {
        int plocal = wv * 16 + v + moff;
        float* op = out + ((size_t)(b * HH + y) * WW + xbase + plocal) * TF + col;
#pragma unroll
        for (int nt = 0; nt < 12; ++nt) op[nt * 16] = acc[nt][v] + bias[nt * 16 + col];
    }
}

// ------- f1 = leaky(conv5x5(x,w_i2f) + conv5x5(h,w_h2f)), all-WMMA, bf16 out -------
__global__ void __launch_bounds__(64)
f1_kernel(const float* __restrict__ x, const float* __restrict__ hprev,
          const __bf16* __restrict__ wh2f_pk, const __bf16* __restrict__ wi2f_pk,
          const float* __restrict__ bi2f, const float* __restrict__ bh2f,
          __bf16* __restrict__ f1out) {
    __shared__ __align__(16) __bf16 hwin[5][WCOLS][FF];     // 23.0 KB shifted window
    __shared__ __align__(16) __bf16 xim[TPX][224];          // 14.0 KB im2col rows
    int wg = blockIdx.x;                                    // B*H*3
    int xseg = wg % 3;
    int rem = wg / 3;
    int y = rem % HH, b = rem / HH;
    int xbase = xseg * TPX;
    int tid = threadIdx.x;

    for (int i = tid; i < 5 * WCOLS * FF; i += 64) {
        int c = i % FF;
        int t2 = i / FF;
        int wc = t2 % WCOLS, dy = t2 / WCOLS;
        int gy = y + dy - 2, gx = xbase + wc - 2;
        float v = 0.f;
        if (gy >= 0 && gy < HH && gx >= 0 && gx < WW)
            v = hprev[((size_t)(b * HH + gy) * WW + gx) * FF + c];
        hwin[dy][wc][c] = (__bf16)v;
    }
    for (int i = tid; i < TPX * 224; i += 64) {
        int px = i / 224, kx = i % 224;
        float v = 0.f;
        if (kx < 200) {
            int tap = kx >> 3, ci = kx & 7;
            int gy = y + tap / 5 - 2, gx = xbase + px + tap % 5 - 2;
            if (gy >= 0 && gy < HH && gx >= 0 && gx < WW)
                v = x[((size_t)(b * HH + gy) * WW + gx) * CC + ci];
        }
        xim[px][kx] = (__bf16)v;
    }
    __syncthreads();

    int wv = tid >> 5, lane = tid & 31;
    int r = lane & 15, kb = (lane >> 4) * 16;
    v8f acc0 = {}, acc1 = {};
    // h2f part: 25 taps x 2 K-chunks
    for (int tap = 0; tap < 25; ++tap) {
        int ty = tap / 5, tx = tap % 5;
        int wc = wv * 16 + r + tx;
        for (int kc = 0; kc < 2; ++kc) {
            v16bf a = load_a_frag(&hwin[ty][wc][kc * 32 + kb]);
            const __bf16* bp = wh2f_pk + (size_t)((tap * 2 + kc) * 2) * 512 + lane * 16;
            v16bf b0 = *(const v16bf*)bp;
            v16bf b1 = *(const v16bf*)(bp + 512);
            acc0 = __builtin_amdgcn_wmma_f32_16x16x32_bf16(false, a, false, b0,
                                                           (short)0, acc0, false, false);
            acc1 = __builtin_amdgcn_wmma_f32_16x16x32_bf16(false, a, false, b1,
                                                           (short)0, acc1, false, false);
        }
    }
    // i2f part: 7 zero-padded K-chunks
    for (int c = 0; c < 7; ++c) {
        v16bf a = load_a_frag(&xim[wv * 16 + r][c * 32 + kb]);
        const __bf16* bp = wi2f_pk + (size_t)(c * 2) * 512 + lane * 16;
        v16bf b0 = *(const v16bf*)bp;
        v16bf b1 = *(const v16bf*)(bp + 512);
        acc0 = __builtin_amdgcn_wmma_f32_16x16x32_bf16(false, a, false, b0,
                                                       (short)0, acc0, false, false);
        acc1 = __builtin_amdgcn_wmma_f32_16x16x32_bf16(false, a, false, b1,
                                                       (short)0, acc1, false, false);
    }
    int col = lane & 15, moff = (lane >> 4) * 8;
#pragma unroll
    for (int v = 0; v < 8; ++v) {
        int plocal = wv * 16 + v + moff;
        __bf16* op = f1out + ((size_t)(b * HH + y) * WW + xbase + plocal) * 32;
        op[col]      = (__bf16)leaky(acc0[v] + bi2f[col] + bh2f[col]);
        op[16 + col] = (__bf16)leaky(acc1[v] + bi2f[16 + col] + bh2f[16 + col]);
    }
}

// ---------------- flows = conv5x5(f1) -> 10 ch via WMMA (N padded to 16) ----------------
__global__ void __launch_bounds__(64)
flows_kernel(const __bf16* __restrict__ f1, const __bf16* __restrict__ wfl_pk,
             const float* __restrict__ bfl, float* __restrict__ flows) {
    __shared__ __align__(16) __bf16 f1win[5][WCOLS][32];    // 11.5 KB
    int wg = blockIdx.x;                                    // B*H*3
    int xseg = wg % 3;
    int rem = wg / 3;
    int y = rem % HH, b = rem / HH;
    int xbase = xseg * TPX;
    int tid = threadIdx.x;

    for (int i = tid; i < 5 * WCOLS * 32; i += 64) {
        int c = i % 32;
        int t2 = i / 32;
        int wc = t2 % WCOLS, dy = t2 / WCOLS;
        int gy = y + dy - 2, gx = xbase + wc - 2;
        __bf16 v = (__bf16)0.f;
        if (gy >= 0 && gy < HH && gx >= 0 && gx < WW)
            v = f1[((size_t)(b * HH + gy) * WW + gx) * 32 + c];
        f1win[dy][wc][c] = v;
    }
    __syncthreads();

    int wv = tid >> 5, lane = tid & 31;
    int r = lane & 15, kb = (lane >> 4) * 16;
    v8f acc = {};
    for (int tap = 0; tap < 25; ++tap) {
        int ty = tap / 5, tx = tap % 5;
        v16bf a = load_a_frag(&f1win[ty][wv * 16 + r + tx][kb]);
        v16bf bfr = *(const v16bf*)(wfl_pk + (size_t)tap * 512 + lane * 16);
        acc = __builtin_amdgcn_wmma_f32_16x16x32_bf16(false, a, false, bfr,
                                                      (short)0, acc, false, false);
    }
    int col = lane & 15, moff = (lane >> 4) * 8;
    if (col < 10) {
        float bv = bfl[col];
#pragma unroll
        for (int v = 0; v < 8; ++v) {
            int plocal = wv * 16 + v + moff;
            size_t p = (size_t)(b * HH + y) * WW + xbase + plocal;
            flows[p * 10 + col] = acc[v] + bv;
        }
    }
}

// ---------------- fused step: warp + 320x192 WMMA GEMM + GRU gates ----------------
__global__ void __launch_bounds__(128)
step_kernel(const float* __restrict__ flows, const float* __restrict__ hprev,
            const float* __restrict__ i2h, const __bf16* __restrict__ wret_pk,
            const float* __restrict__ bret, float* __restrict__ hnext) {
    __shared__ __align__(16) __bf16 Atile[16][KH];          // 10 KB
    int pbase = blockIdx.x * 16;
    int tid = threadIdx.x;

    // build warped A-tile (bilinear gather, convert to bf16)
    {
        int px = tid >> 3;          // 0..15
        int cpart = tid & 7;        // 8-channel slice
        int p = pbase + px;
        int b = p / (HH * WW);
        const float* fl = flows + (size_t)p * 10;
        const float* img = hprev + (size_t)b * HH * WW * FF;
        for (int l = 0; l < LL; ++l) {
            float fx = -fl[2 * l];
            float fy = -fl[2 * l + 1];
            float xg = 0.5f * ((fx + 1.f) * (float)(WW - 2));
            float yg = 0.5f * ((fy + 1.f) * (float)(HH - 2));
            int x0 = (int)floorf(xg), y0 = (int)floorf(yg);
            int x0c = clampi(x0, 0, WW - 1), x1c = clampi(x0 + 1, 0, WW - 1);
            int y0c = clampi(y0, 0, HH - 1), y1c = clampi(y0 + 1, 0, HH - 1);
            float x0f = (float)x0c, x1f = (float)x1c;
            float y0f = (float)y0c, y1f = (float)y1c;
            float wa = (x1f - xg) * (y1f - yg);
            float wb = (x1f - xg) * (yg - y0f);
            float wc = (xg - x0f) * (y1f - yg);
            float wd = (xg - x0f) * (yg - y0f);
            const float* Ia = img + ((size_t)y0c * WW + x0c) * FF;
            const float* Ib = img + ((size_t)y1c * WW + x0c) * FF;
            const float* Ic = img + ((size_t)y0c * WW + x1c) * FF;
            const float* Id = img + ((size_t)y1c * WW + x1c) * FF;
#pragma unroll
            for (int j = 0; j < 8; ++j) {
                int c = cpart * 8 + j;
                float v = wa * Ia[c] + wb * Ib[c] + wc * Ic[c] + wd * Id[c];
                Atile[px][l * FF + c] = (__bf16)v;
            }
        }
    }
    __syncthreads();

    // wave w handles N-tiles {w, w+4, w+8} = (r,u,m) for channels 16w..16w+15
    int wv = tid >> 5, lane = tid & 31;
    int r = lane & 15, kb = (lane >> 4) * 16;
    v8f accR = {}, accU = {}, accM = {};
    for (int kc = 0; kc < 10; ++kc) {
        v16bf a = load_a_frag(&Atile[r][kc * 32 + kb]);
        const __bf16* bp = wret_pk + (size_t)(kc * 12) * 512 + lane * 16;
        v16bf bR = *(const v16bf*)(bp + (size_t)wv * 512);
        v16bf bU = *(const v16bf*)(bp + (size_t)(wv + 4) * 512);
        v16bf bM = *(const v16bf*)(bp + (size_t)(wv + 8) * 512);
        accR = __builtin_amdgcn_wmma_f32_16x16x32_bf16(false, a, false, bR,
                                                       (short)0, accR, false, false);
        accU = __builtin_amdgcn_wmma_f32_16x16x32_bf16(false, a, false, bU,
                                                       (short)0, accU, false, false);
        accM = __builtin_amdgcn_wmma_f32_16x16x32_bf16(false, a, false, bM,
                                                       (short)0, accM, false, false);
    }

    int col = lane & 15, moff = (lane >> 4) * 8;
    int ch = wv * 16 + col;
    float brv = bret[ch], buv = bret[FF + ch], bmv = bret[2 * FF + ch];
#pragma unroll
    for (int v = 0; v < 8; ++v) {
        int p = pbase + v + moff;
        const float* ip = i2h + (size_t)p * TF;
        float h2r = accR[v] + brv;
        float h2u = accU[v] + buv;
        float h2m = accM[v] + bmv;
        float rg = sigmoidf(ip[ch] + h2r);
        float ug = sigmoidf(ip[FF + ch] + h2u);
        float mm = leaky(ip[2 * FF + ch] + rg * h2m);
        float hp = hprev[(size_t)p * FF + ch];
        hnext[(size_t)p * FF + ch] = ug * hp + (1.f - ug) * mm;
    }
}

// ---------------- host orchestration ----------------
extern "C" void kernel_launch(void* const* d_in, const int* in_sizes, int n_in,
                              void* d_out, int out_size, void* d_ws, size_t ws_size,
                              hipStream_t stream) {
    const float* inputs  = (const float*)d_in[0];
    const float* w_i2h   = (const float*)d_in[1];
    const float* b_i2h   = (const float*)d_in[2];
    const float* w_i2f   = (const float*)d_in[3];
    const float* b_i2f   = (const float*)d_in[4];
    const float* w_h2f   = (const float*)d_in[5];
    const float* b_h2f   = (const float*)d_in[6];
    const float* w_flows = (const float*)d_in[7];
    const float* b_flows = (const float*)d_in[8];
    const float* w_ret   = (const float*)d_in[9];
    const float* b_ret   = (const float*)d_in[10];
    float* outs = (float*)d_out;

    char* ws = (char*)d_ws;
    size_t off = 0;
    float* zero_h = (float*)(ws + off);    off += (size_t)NPIX * FF * 4;   // h0
    float* i2h_t  = (float*)(ws + off);    off += (size_t)NPIX * TF * 4;
    __bf16* f1buf = (__bf16*)(ws + off);   off += (size_t)NPIX * 32 * 2;
    float* flbuf  = (float*)(ws + off);    off += (size_t)NPIX * 10 * 4;
    __bf16* wret_pk = (__bf16*)(ws + off); off += (size_t)120 * 512 * 2;
    __bf16* wh2f_pk = (__bf16*)(ws + off); off += (size_t)100 * 512 * 2;
    __bf16* wi2f_pk = (__bf16*)(ws + off); off += (size_t)14 * 512 * 2;
    __bf16* wfl_pk  = (__bf16*)(ws + off); off += (size_t)25 * 512 * 2;
    __bf16* wi2h_pk = (__bf16*)(ws + off); off += (size_t)36 * 512 * 2;

    size_t nz = (size_t)NPIX * FF;
    fill_zero<<<(int)((nz + 255) / 256), 256, 0, stream>>>(zero_h, nz);
    pack_wret<<<(120 * 32 + 255) / 256, 256, 0, stream>>>(w_ret, wret_pk);
    pack_wh2f<<<(100 * 32 + 255) / 256, 256, 0, stream>>>(w_h2f, wh2f_pk);
    pack_wi2f<<<(14 * 32 + 255) / 256, 256, 0, stream>>>(w_i2f, wi2f_pk);
    pack_wfl<<<(25 * 32 + 255) / 256, 256, 0, stream>>>(w_flows, wfl_pk);
    pack_wi2h<<<(36 * 32 + 255) / 256, 256, 0, stream>>>(w_i2h, wi2h_pk);

    size_t hstride = (size_t)NPIX * FF;
    for (int t = 0; t < SS; ++t) {
        const float* x_t = inputs + (size_t)t * NPIX * CC;
        const float* hprev = (t == 0) ? zero_h : (outs + (size_t)(t - 1) * hstride);
        float* hnext = outs + (size_t)t * hstride;
        i2h_kernel<<<BB * HH * 3, 64, 0, stream>>>(x_t, wi2h_pk, b_i2h, i2h_t);
        f1_kernel<<<BB * HH * 3, 64, 0, stream>>>(x_t, hprev, wh2f_pk, wi2f_pk,
                                                  b_i2f, b_h2f, f1buf);
        flows_kernel<<<BB * HH * 3, 64, 0, stream>>>(f1buf, wfl_pk, b_flows, flbuf);
        step_kernel<<<NPIX / 16, 128, 0, stream>>>(flbuf, hprev, i2h_t, wret_pk,
                                                   b_ret, hnext);
    }
    // last_h = outs[S-1]
    hipMemcpyAsync(outs + (size_t)SS * hstride, outs + (size_t)(SS - 1) * hstride,
                   hstride * sizeof(float), hipMemcpyDeviceToDevice, stream);
}
